// TorchGroupQueryAttention_86535001080133
// MI455X (gfx1250) — matrix operand order repaired and checked by reading, not compiled, not run on hardware
//
#include <hip/hip_runtime.h>
#include <hip/hip_bf16.h>

// ---------------------------------------------------------------------------
// GQA forward for MI455X (gfx1250, wave32).
//   - all GEMMs via v_wmma_f32_16x16x32_bf16 (bf16 in, fp32 accum)
//   - matrix tiles staged LDS-side by the Tensor Data Mover (TDM),
//     double-buffered, tracked with TENSORcnt
// ---------------------------------------------------------------------------

typedef __attribute__((ext_vector_type(16))) __bf16 v16bf;
typedef __attribute__((ext_vector_type(8)))  __bf16 v8bf;
typedef __attribute__((ext_vector_type(4)))  __bf16 v4bf;
typedef __attribute__((ext_vector_type(8)))  float  v8f;
typedef __attribute__((ext_vector_type(4)))  float  v4f;
typedef __attribute__((ext_vector_type(4)))  unsigned int u32x4;
typedef __attribute__((ext_vector_type(8)))  int    i32x8;
typedef __attribute__((ext_vector_type(4)))  int    i32x4;

#define LDS_STRIDE 48   // bf16 elems per LDS row: 32 data + 16 pad (96 B)

#if defined(__HIP_DEVICE_COMPILE__) && \
    __has_builtin(__builtin_amdgcn_tensor_load_to_lds) && \
    __has_builtin(__builtin_amdgcn_s_wait_tensorcnt)
#define USE_TDM 1
#else
#define USE_TDM 0
#endif

// ---------------------------------------------------------------------------
// fp32 -> bf16 conversion, 4 elements / thread
// ---------------------------------------------------------------------------
__global__ __launch_bounds__(256)
void cvt_bf16(const float* __restrict__ in, __bf16* __restrict__ out, int n4)
{
    int i = blockIdx.x * blockDim.x + threadIdx.x;
    if (i < n4) {
        v4f f = ((const v4f*)in)[i];
        v4bf o;
#pragma unroll
        for (int k = 0; k < 4; ++k) o[k] = (__bf16)f[k];
        ((v4bf*)out)[i] = o;
    }
}

#if USE_TDM
// ---------------------------------------------------------------------------
// TDM: async-copy a [rows x cols] bf16 tile (row stride = strideElems) from
// global into LDS at byte offset ldsOff, inserting 32B of padding after every
// 64B row so LDS rows land at LDS_STRIDE(=48) bf16 granularity.
// D# bitfields per CDNA5 ISA ch.8 (group0: count/lds/global/type,
// group1: data_size|pad cfg, tensor dims, tile dims, dim0 stride).
// ---------------------------------------------------------------------------
__device__ __forceinline__ void tdm_load_tile(const __bf16* gsrc, unsigned ldsOff,
                                              int rows, int cols, int strideElems)
{
    unsigned long long ga = (unsigned long long)(size_t)gsrc;
    u32x4 g0;
    g0[0] = 1u;                                        // count=1, user mode
    g0[1] = ldsOff;                                    // lds_addr (bytes)
    g0[2] = (unsigned)ga;                              // global_addr[31:0]
    g0[3] = (unsigned)((ga >> 32) & 0x01FFFFFFull)     // global_addr[56:32]
          | (2u << 30);                                // type = 2 ("image")
    i32x8 g1;
    g1[0] = (1 << 16)        // data_size = 2 bytes
          | (1 << 20)        // pad_enable
          | (3 << 22)        // pad_interval: 16 DWORDs (64 B)
          | (7 << 25);       // pad_amount:   8 DWORDs (32 B)
    g1[1] = (cols & 0xFFFF) << 16;                         // tensor_dim0[15:0]
    g1[2] = (((unsigned)cols >> 16) & 0xFFFF)              // tensor_dim0[31:16]
          | ((rows & 0xFFFF) << 16);                       // tensor_dim1[15:0]
    g1[3] = (((unsigned)rows >> 16) & 0xFFFF)              // tensor_dim1[31:16]
          | ((cols & 0xFFFF) << 16);                       // tile_dim0
    g1[4] = rows & 0xFFFF;                                 // tile_dim1 (tile_dim2=0)
    g1[5] = strideElems;                                   // tensor_dim0_stride[31:0]
    g1[6] = 0;
    g1[7] = 0;
    i32x4 z4 = {0, 0, 0, 0};
#if __clang_major__ >= 23
    i32x8 z8 = {0, 0, 0, 0, 0, 0, 0, 0};
    __builtin_amdgcn_tensor_load_to_lds(g0, g1, z4, z4, z8, 0);
#else
    __builtin_amdgcn_tensor_load_to_lds(g0, g1, z4, z4, 0);
#endif
}

__device__ __forceinline__ unsigned lds_off(const void* p)
{
    return (unsigned)(size_t)p;     // LDS aperture lives in addr[63:32]
}
#endif // USE_TDM

// ---------------------------------------------------------------------------
// One BK=32 compute step: 2x2 grid of v_wmma_f32_16x16x32_bf16 per wave.
// A lane m=frm holds K = hi*16..hi*16+15 (contiguous);
// B lane col n=frm holds K = hi*16..hi*16+15 (contiguous).
// ---------------------------------------------------------------------------
__device__ __forceinline__ void wmma_step(const __bf16* __restrict__ lA,
                                          const __bf16* __restrict__ lB,
                                          int wm, int wn, int frm, int hi,
                                          v8f (&acc)[2][2])
{
    v16bf afrag[2];
#pragma unroll
    for (int i = 0; i < 2; ++i) {
        const __bf16* pa = &lA[(wm + i * 16 + frm) * LDS_STRIDE + hi * 16];
        v8bf a0 = *(const v8bf*)pa;
        v8bf a1 = *(const v8bf*)(pa + 8);
#pragma unroll
        for (int e = 0; e < 8; ++e) { afrag[i][e] = a0[e]; afrag[i][8 + e] = a1[e]; }
    }
    v16bf bfrag[2];
#pragma unroll
    for (int j = 0; j < 2; ++j) {
        const __bf16* pb = &lB[(wn + j * 16 + frm) * LDS_STRIDE + hi * 16];
        v8bf b0 = *(const v8bf*)pb;
        v8bf b1 = *(const v8bf*)(pb + 8);
#pragma unroll
        for (int e = 0; e < 8; ++e) { bfrag[j][e] = b0[e]; bfrag[j][8 + e] = b1[e]; }
    }
#pragma unroll
    for (int i = 0; i < 2; ++i)
#pragma unroll
        for (int j = 0; j < 2; ++j)
            acc[i][j] = __builtin_amdgcn_wmma_f32_16x16x32_bf16(
                false, afrag[i], false, bfrag[j], (short)0, acc[i][j], false, false);
}

// ---------------------------------------------------------------------------
// Tiled WMMA GEMM:  C[z] = alpha * A[z] (MxK, K-major) * B[z]^T (NxK, K-major)
//   A_IS_F32: A is fp32 in global, converted to bf16 on the LDS-fill path
//             (manual staging; TDM cannot convert)
//   OUT_MODE: 0 = fp32 C,  1 = bf16 C,  2 = bf16 C stored transposed
//   per-z offsets: base += (z >> shift) * stride   (elements)
// Block: 128 threads = 4 waves; 64x64 tile; wave = 32x32 sub-tile (2x2 WMMA).
// bf16-A path: TDM double-buffered staging, TENSORcnt-tracked.
// ---------------------------------------------------------------------------
template<bool A_IS_F32, int OUT_MODE>
__global__ __launch_bounds__(128)
void gemm_wmma(const void* __restrict__ Av, const __bf16* __restrict__ B,
               void* __restrict__ Cv,
               int K, int lda, int ldb, int ldc, float alpha,
               unsigned long long strideA, int shiftA,
               unsigned long long strideB, int shiftB,
               unsigned long long strideC, int shiftC)
{
    __shared__ __bf16 ldsA[2][64 * LDS_STRIDE];
    __shared__ __bf16 ldsB[2][64 * LDS_STRIDE];

    const int tid  = threadIdx.x;
    const int lane = tid & 31;
    const int wave = tid >> 5;
    const int wm   = (wave >> 1) * 32;
    const int wn   = (wave & 1)  * 32;

    const int z      = blockIdx.z;
    const int blockN = blockIdx.x * 64;
    const int blockM = blockIdx.y * 64;

    const size_t aOff = (size_t)(z >> shiftA) * strideA;
    const size_t bOff = (size_t)(z >> shiftB) * strideB;
    const size_t cOff = (size_t)(z >> shiftC) * strideC;

    const float*  Af = (const float*)Av  + aOff;
    const __bf16* Ab = (const __bf16*)Av + aOff;
    const __bf16* Bb = B + bOff;

    v8f acc[2][2] = {};

    const int frm = lane & 15;
    const int hi  = lane >> 4;

#if USE_TDM
    if constexpr (!A_IS_F32) {
        // ---------------- TDM double-buffered pipeline ----------------
        if (wave == 0) {
            tdm_load_tile(Ab + (size_t)blockM * lda, lds_off(&ldsA[0][0]), 64, 32, lda);
            tdm_load_tile(Bb + (size_t)blockN * ldb, lds_off(&ldsB[0][0]), 64, 32, ldb);
        }
        int cur = 0;
        for (int k0 = 0; k0 < K; k0 += 32) {
            const int nxt = cur ^ 1;
            if (wave == 0) {
                if (k0 + 32 < K) {
                    tdm_load_tile(Ab + (size_t)blockM * lda + k0 + 32,
                                  lds_off(&ldsA[nxt][0]), 64, 32, lda);
                    tdm_load_tile(Bb + (size_t)blockN * ldb + k0 + 32,
                                  lds_off(&ldsB[nxt][0]), 64, 32, ldb);
                    __builtin_amdgcn_s_wait_tensorcnt(2);  // current 2 done
                } else {
                    __builtin_amdgcn_s_wait_tensorcnt(0);
                }
            }
            __syncthreads();                 // current buffer visible to all waves
            wmma_step(&ldsA[cur][0], &ldsB[cur][0], wm, wn, frm, hi, acc);
            __syncthreads();                 // all reads done before buffer reuse
            cur = nxt;
        }
    } else
#endif
    {
        // ---------------- manual staging (fp32 A converts to bf16) ----------
        const int lr = tid >> 2;          // 0..31 : row within 32-row pass
        const int lc = (tid & 3) * 8;     // 0,8,16,24

        for (int k0 = 0; k0 < K; k0 += 32) {
#pragma unroll
            for (int t = 0; t < 2; ++t) {
                const int row = lr + t * 32;
                __bf16* dst = &ldsA[0][row * LDS_STRIDE + lc];
                if (A_IS_F32) {
                    const float* src = Af + (size_t)(blockM + row) * lda + k0 + lc;
                    v4f f0 = *(const v4f*)(src);
                    v4f f1 = *(const v4f*)(src + 4);
                    v8bf o;
#pragma unroll
                    for (int i = 0; i < 4; ++i) { o[i] = (__bf16)f0[i]; o[4 + i] = (__bf16)f1[i]; }
                    *(v8bf*)dst = o;
                } else {
                    const __bf16* src = Ab + (size_t)(blockM + row) * lda + k0 + lc;
                    *(v8bf*)dst = *(const v8bf*)src;
                }
            }
#pragma unroll
            for (int t = 0; t < 2; ++t) {
                const int row = lr + t * 32;
                const __bf16* src = Bb + (size_t)(blockN + row) * ldb + k0 + lc;
                *(v8bf*)&ldsB[0][row * LDS_STRIDE + lc] = *(const v8bf*)src;
            }
            __syncthreads();
            wmma_step(&ldsA[0][0], &ldsB[0][0], wm, wn, frm, hi, acc);
            __syncthreads();
        }
    }

    // ---- store:  C VGPR r -> row r + 8*hi, col frm ----
#pragma unroll
    for (int i = 0; i < 2; ++i) {
#pragma unroll
        for (int j = 0; j < 2; ++j) {
            const int col = blockN + wn + j * 16 + frm;
#pragma unroll
            for (int r = 0; r < 8; ++r) {
                const int row = blockM + wm + i * 16 + r + hi * 8;
                const float val = acc[i][j][r] * alpha;
                if (OUT_MODE == 0) {
                    ((float*)Cv + cOff)[(size_t)row * ldc + col] = val;
                } else if (OUT_MODE == 1) {
                    ((__bf16*)Cv + cOff)[(size_t)row * ldc + col] = (__bf16)val;
                } else {
                    ((__bf16*)Cv + cOff)[(size_t)col * ldc + row] = (__bf16)val;
                }
            }
        }
    }
}

// ---------------------------------------------------------------------------
// Row softmax over 2048 columns, in place, with additive attention mask.
// grid = (S rows, H heads), 256 threads, 8 columns / thread.
// ---------------------------------------------------------------------------
__global__ __launch_bounds__(256)
void softmax_rows(float* __restrict__ w, const float* __restrict__ mask)
{
    constexpr int S = 2048;
    const int row = blockIdx.x;
    const int h   = blockIdx.y;
    float* p = w + ((size_t)h * S + row) * S;
    const int t = threadIdx.x;

    float vals[8];
    float mx = -3.4e38f;
#pragma unroll
    for (int i = 0; i < 8; ++i) {
        const int c = t + i * 256;
        const float s = p[c] + (1.0f - mask[c]) * -10000.0f;
        vals[i] = s;
        mx = fmaxf(mx, s);
    }

    __shared__ float red[256];
    red[t] = mx;
    __syncthreads();
    for (int off = 128; off > 0; off >>= 1) {
        if (t < off) red[t] = fmaxf(red[t], red[t + off]);
        __syncthreads();
    }
    mx = red[0];
    __syncthreads();

    float sum = 0.f;
#pragma unroll
    for (int i = 0; i < 8; ++i) { vals[i] = __expf(vals[i] - mx); sum += vals[i]; }
    red[t] = sum;
    __syncthreads();
    for (int off = 128; off > 0; off >>= 1) {
        if (t < off) red[t] += red[t + off];
        __syncthreads();
    }
    const float inv = 1.0f / red[0];
#pragma unroll
    for (int i = 0; i < 8; ++i) p[t + i * 256] = vals[i] * inv;
}

// ---------------------------------------------------------------------------
extern "C" void kernel_launch(void* const* d_in, const int* in_sizes, int n_in,
                              void* d_out, int out_size, void* d_ws, size_t ws_size,
                              hipStream_t stream)
{
    (void)in_sizes; (void)n_in; (void)out_size; (void)ws_size;

    const float* x    = (const float*)d_in[0];
    const float* mask = (const float*)d_in[1];
    // d_in[2] = position_ids (unused by reference)
    const float* wq   = (const float*)d_in[3];
    const float* wk   = (const float*)d_in[4];
    const float* wv   = (const float*)d_in[5];
    const float* wo   = (const float*)d_in[6];

    constexpr int S = 2048, E = 2048, H = 32, D = 64, KVE = 512;
    constexpr size_t SE  = (size_t)S * E;     // 4194304
    constexpr size_t SKV = (size_t)S * KVE;   // 1048576

    // workspace layout (bf16 buffers), 48 MiB total
    char* ws = (char*)d_ws;
    __bf16* xb  = (__bf16*)ws; ws += SE  * 2;
    __bf16* wqb = (__bf16*)ws; ws += SE  * 2;
    __bf16* wkb = (__bf16*)ws; ws += SKV * 2;
    __bf16* wvb = (__bf16*)ws; ws += SKV * 2;
    __bf16* wob = (__bf16*)ws; ws += SE  * 2;
    __bf16* Qb  = (__bf16*)ws; ws += SE  * 2;   // scaled by 1/sqrt(D)
    __bf16* Kb  = (__bf16*)ws; ws += SKV * 2;   // [S x KVE]
    __bf16* Vt  = (__bf16*)ws; ws += SKV * 2;   // transposed: [KVE x S]
    __bf16* Ob  = (__bf16*)ws; ws += SE  * 2;   // attention output [S x E]

    float* out = (float*)d_out;      // [S, E]
    float* wts = out + SE;           // [H, S, S] softmax weights (output #2)

    // 1) fp32 -> bf16
    cvt_bf16<<<(int)(SE  / 4 / 256), 256, 0, stream>>>(x,  xb,  (int)(SE  / 4));
    cvt_bf16<<<(int)(SE  / 4 / 256), 256, 0, stream>>>(wq, wqb, (int)(SE  / 4));
    cvt_bf16<<<(int)(SKV / 4 / 256), 256, 0, stream>>>(wk, wkb, (int)(SKV / 4));
    cvt_bf16<<<(int)(SKV / 4 / 256), 256, 0, stream>>>(wv, wvb, (int)(SKV / 4));
    cvt_bf16<<<(int)(SE  / 4 / 256), 256, 0, stream>>>(wo, wob, (int)(SE  / 4));

    const dim3 blk(128);

    // 2) Q = (x @ wq^T) * 1/sqrt(D)   -> bf16 [S x E]
    gemm_wmma<false, 1><<<dim3(E / 64, S / 64, 1), blk, 0, stream>>>(
        xb, wqb, Qb, E, E, E, E, 0.125f, 0ull, 0, 0ull, 0, 0ull, 0);

    // 3) K = x @ wk^T                 -> bf16 [S x KVE]
    gemm_wmma<false, 1><<<dim3(KVE / 64, S / 64, 1), blk, 0, stream>>>(
        xb, wkb, Kb, E, E, E, KVE, 1.0f, 0ull, 0, 0ull, 0, 0ull, 0);

    // 4) V = x @ wv^T, stored transposed -> bf16 Vt [KVE x S]
    gemm_wmma<false, 2><<<dim3(KVE / 64, S / 64, 1), blk, 0, stream>>>(
        xb, wvb, Vt, E, E, E, S, 1.0f, 0ull, 0, 0ull, 0, 0ull, 0);

    // 5) per-head scores = Q_h @ K_{h/4}^T -> fp32 straight into weights region
    gemm_wmma<false, 0><<<dim3(S / 64, S / 64, H), blk, 0, stream>>>(
        Qb, Kb, wts, D, E, KVE, S, 1.0f,
        (unsigned long long)D, 0,            // A: col block h*64 of Qb
        (unsigned long long)D, 2,            // B: col block (h>>2)*64 of Kb
        (unsigned long long)S * S, 0);       // C: head slab of weights

    // 6) softmax rows (mask applied), in place
    softmax_rows<<<dim3(S, H), 256, 0, stream>>>(wts, mask);

    // 7) per-head O_h = P_h @ V_{h/4}  (A fp32->bf16 on load; B = Vt rows)
    gemm_wmma<true, 1><<<dim3(D / 64, S / 64, H), blk, 0, stream>>>(
        wts, Vt, Ob, S, S, S, E, 1.0f,
        (unsigned long long)S * S, 0,        // A: head slab of weights
        (unsigned long long)D * S, 2,        // B: rows (h>>2)*64.. of Vt
        (unsigned long long)D, 0);           // C: column block h*64 of Ob

    // 8) out = O @ wo^T -> fp32 d_out
    gemm_wmma<false, 0><<<dim3(E / 64, S / 64, 1), blk, 0, stream>>>(
        Ob, wob, out, E, E, E, E, 1.0f, 0ull, 0, 0ull, 0, 0ull, 0);
}
